// ColPali2Loss_64965675320062
// MI455X (gfx1250) — matrix operand order repaired and checked by reading, not compiled, not run on hardware
//
#include <hip/hip_runtime.h>
#include <hip/hip_bf16.h>

typedef __attribute__((ext_vector_type(16))) __bf16 v16bf;
typedef __attribute__((ext_vector_type(8)))  float  v8f;
typedef __attribute__((ext_vector_type(4)))  unsigned int u32x4;
typedef unsigned short u16b;

#define BB   64
#define NQ   32
#define NS   1024
#define DD   128
#define DV   1024

union Frag {
    u32x4 q[2];
    v16bf b;
};

// ---------------------------------------------------------------------------
// Kernel 1: split f32 -> (hi, lo) bf16 and scatter into WMMA fragment-swizzled
// layout so the GEMM kernel's loads are fully coalesced.
//
// B (d_multi) fragment block for (c, ntile, ks): 32 lanes x 32 bytes = 1 KB.
//   lane = (s & 15) + 16 * ((k & 31) >> 4), elem e = k & 15   (contiguous k order)
// A (q_multi) fragment block for (b, mtile, ks): 32 lanes x 32 bytes = 1 KB.
//   lane = (m & 15) + 16 * ((k & 31) >> 4), elem e = k & 15
// ---------------------------------------------------------------------------
__device__ __forceinline__ u16b f32_to_bf16_rne(float x) {
    unsigned u = __float_as_uint(x);
    unsigned r = u + 0x7FFFu + ((u >> 16) & 1u);
    return (u16b)(r >> 16);
}

__global__ __launch_bounds__(256) void convert_kernel(
        const float* __restrict__ qm, const float* __restrict__ dm,
        u16b* __restrict__ qh, u16b* __restrict__ ql,
        u16b* __restrict__ dh, u16b* __restrict__ dl) {
    const size_t DN = (size_t)BB * NS * DD;   // 8388608
    const size_t QN = (size_t)BB * NQ * DD;   // 262144
    const size_t total = DN + QN;
    const size_t stride = (size_t)gridDim.x * blockDim.x;
    for (size_t idx = (size_t)blockIdx.x * blockDim.x + threadIdx.x;
         idx < total; idx += stride) {
        float x;
        u16b *ph, *pl;
        size_t dst;
        if (idx < DN) {
            int c = (int)(idx / (NS * DD));
            int rem = (int)(idx % (NS * DD));
            int s = rem / DD, k = rem % DD;
            int nt = s >> 4, l15 = s & 15;
            int ks = k >> 5, r = k & 31;
            int lane = l15 + 16 * (r >> 4);
            int e = r & 15;
            dst = ((((size_t)c * 64 + nt) * 4 + ks) * 32 + lane) * 16 + e;
            x = dm[idx]; ph = dh; pl = dl;
        } else {
            size_t qi = idx - DN;
            int b = (int)(qi / (NQ * DD));
            int rem = (int)(qi % (NQ * DD));
            int m = rem / DD, k = rem % DD;
            int mt = m >> 4, l15 = m & 15;
            int ks = k >> 5, r = k & 31;
            int lane = l15 + 16 * (r >> 4);
            int e = r & 15;
            dst = ((((size_t)b * 2 + mt) * 4 + ks) * 32 + lane) * 16 + e;
            x = qm[qi]; ph = qh; pl = ql;
        }
        u16b hb = f32_to_bf16_rne(x);
        float hf = __uint_as_float(((unsigned)hb) << 16);
        u16b lb = f32_to_bf16_rne(x - hf);
        ph[dst] = hb;
        pl[dst] = lb;
    }
}

// ---------------------------------------------------------------------------
// Kernel 2: per (b,c) pair: S = q[b](32x128) x d[c]^T(128x1024) via bf16 WMMA
// hi/lo split (qh*dh + qh*dl + ql*dh). Each wave computes BOTH m-tiles for its
// 8 n-tiles, so each (bh, bl) fragment pair feeds 6 WMMAs and d[c] is loaded once/block.
// Rowwise max over s, sum over n -> scores_mv[b][c].
// ---------------------------------------------------------------------------
__device__ __forceinline__ void load_frag(Frag& f, const u16b* p) {
    f.q[0] = *(const u32x4*)(p);
    f.q[1] = *(const u32x4*)(p + 8);
}

__global__ __launch_bounds__(256) void pair_kernel(
        const u16b* __restrict__ qh, const u16b* __restrict__ ql,
        const u16b* __restrict__ dh, const u16b* __restrict__ dl,
        float* __restrict__ scores_mv) {
    const int c = blockIdx.x;
    const int b = blockIdx.y;
    const int tid  = threadIdx.x;
    const int wave = tid >> 5;     // 0..7 : n-tile group
    const int lane = tid & 31;

    __shared__ unsigned rowmax_u[NQ];
    if (tid < NQ) rowmax_u[tid] = 0u;   // below encode(-inf)
    __syncthreads();

    // Preload all A fragments (both m-tiles, hi & lo, 4 k-steps): 128 VGPRs.
    Frag ah[2][4], al[2][4];
#pragma unroll
    for (int mt = 0; mt < 2; ++mt) {
#pragma unroll
        for (int ks = 0; ks < 4; ++ks) {
            const size_t base = ((((size_t)b * 2 + mt) * 4 + ks) * 32 + lane) * 16;
            load_frag(ah[mt][ks], qh + base);
            load_frag(al[mt][ks], ql + base);
        }
    }

    v8f rmax0, rmax1;
#pragma unroll
    for (int r = 0; r < 8; ++r) { rmax0[r] = -1e30f; rmax1[r] = -1e30f; }

    // Wave handles n-tiles: wave, wave+8, ..., wave+56
    for (int t = 0; t < 8; ++t) {
        const int nt = wave + 8 * t;
        v8f acc0, acc1;
#pragma unroll
        for (int r = 0; r < 8; ++r) { acc0[r] = 0.0f; acc1[r] = 0.0f; }

#pragma unroll
        for (int ks = 0; ks < 4; ++ks) {
            const size_t base = ((((size_t)c * 64 + nt) * 4 + ks) * 32 + lane) * 16;
            Frag bh, bl;
            load_frag(bh, dh + base);
            load_frag(bl, dl + base);
            acc0 = __builtin_amdgcn_wmma_f32_16x16x32_bf16(
                       false, ah[0][ks].b, false, bh.b, (short)0, acc0, false, false);
            acc1 = __builtin_amdgcn_wmma_f32_16x16x32_bf16(
                       false, ah[1][ks].b, false, bh.b, (short)0, acc1, false, false);
            acc0 = __builtin_amdgcn_wmma_f32_16x16x32_bf16(
                       false, ah[0][ks].b, false, bl.b, (short)0, acc0, false, false);
            acc1 = __builtin_amdgcn_wmma_f32_16x16x32_bf16(
                       false, ah[1][ks].b, false, bl.b, (short)0, acc1, false, false);
            acc0 = __builtin_amdgcn_wmma_f32_16x16x32_bf16(
                       false, al[0][ks].b, false, bh.b, (short)0, acc0, false, false);
            acc1 = __builtin_amdgcn_wmma_f32_16x16x32_bf16(
                       false, al[1][ks].b, false, bh.b, (short)0, acc1, false, false);
        }
#pragma unroll
        for (int r = 0; r < 8; ++r) {
            rmax0[r] = fmaxf(rmax0[r], acc0[r]);
            rmax1[r] = fmaxf(rmax1[r], acc1[r]);
        }
    }

    // Merge row maxima across lanes & waves (order-preserving uint ds_max).
    // C/D layout: VGPR r, lane<16 -> row r ; lane>=16 -> row r+8 (per m-tile).
    const int half = lane >> 4;
#pragma unroll
    for (int r = 0; r < 8; ++r) {
        const int row = r + 8 * half;
        unsigned b0 = __float_as_uint(rmax0[r]);
        unsigned e0 = (b0 & 0x80000000u) ? ~b0 : (b0 | 0x80000000u);
        atomicMax(&rowmax_u[row], e0);
        unsigned b1 = __float_as_uint(rmax1[r]);
        unsigned e1 = (b1 & 0x80000000u) ? ~b1 : (b1 | 0x80000000u);
        atomicMax(&rowmax_u[16 + row], e1);
    }
    __syncthreads();

    if (wave == 0) {
        unsigned m = rowmax_u[lane];
        float f = (m & 0x80000000u) ? __uint_as_float(m & 0x7FFFFFFFu)
                                    : __uint_as_float(~m);
#pragma unroll
        for (int off = 16; off >= 1; off >>= 1) f += __shfl_xor(f, off, 32);
        if (lane == 0) scores_mv[b * BB + c] = f;
    }
}

// ---------------------------------------------------------------------------
// Kernel 3: single-vector scores row b + logsumexp -> sv_part[b] = lse - diag
// ---------------------------------------------------------------------------
__global__ __launch_bounds__(256) void sv_kernel(
        const float* __restrict__ qs, const float* __restrict__ ds,
        float* __restrict__ sv_part) {
    const int b   = blockIdx.x;
    const int tid = threadIdx.x;
    const int cg  = tid >> 2;    // 0..63 : which column c
    const int sub = tid & 3;     // 4 threads per dot product

    const float* qrow = qs + (size_t)b  * DV;
    const float* drow = ds + (size_t)cg * DV;
    const int k0 = sub * (DV / 4);

    float s = 0.0f;
#pragma unroll 4
    for (int k = 0; k < DV / 4; k += 4) {
        float4 a  = *(const float4*)(qrow + k0 + k);
        float4 dd = *(const float4*)(drow + k0 + k);
        s += a.x * dd.x + a.y * dd.y + a.z * dd.z + a.w * dd.w;
    }
    s += __shfl_xor(s, 1, 32);
    s += __shfl_xor(s, 2, 32);

    __shared__ float row[BB];
    if (sub == 0) row[cg] = s;
    __syncthreads();

    if (tid < 32) {
        const int lane = tid;
        float f0 = row[lane];
        float f1 = row[lane + 32];
        float m = fmaxf(f0, f1);
#pragma unroll
        for (int off = 16; off >= 1; off >>= 1)
            m = fmaxf(m, __shfl_xor(m, off, 32));
        float e = __expf(f0 - m) + __expf(f1 - m);
#pragma unroll
        for (int off = 16; off >= 1; off >>= 1) e += __shfl_xor(e, off, 32);
        if (lane == 0) {
            float lse = m + __logf(e);
            sv_part[b] = lse - row[b];
        }
    }
}

// ---------------------------------------------------------------------------
// Kernel 4: finalize: mv = mean(softplus(neg-pos)); out = 0.5*sv + 0.5*mv
// ---------------------------------------------------------------------------
__global__ __launch_bounds__(64) void finalize_kernel(
        const float* __restrict__ scores_mv, const float* __restrict__ sv_part,
        float* __restrict__ out) {
    __shared__ float sh[BB];
    const int t = threadIdx.x;   // 0..63

    const float* rowp = scores_mv + t * BB;
    float pos = rowp[t];
    float neg = -1e30f;
    for (int c = 0; c < BB; ++c) {
        float v = rowp[c];
        if (c != t) neg = fmaxf(neg, v);
    }
    float x  = neg - pos;
    float sp = (x > 0.0f) ? x + log1pf(expf(-x)) : log1pf(expf(x));

    sh[t] = 0.5f * sv_part[t] + (0.5f / (float)BB) * sp;
    __syncthreads();
    for (int s = 32; s >= 1; s >>= 1) {
        if (t < s) sh[t] += sh[t + s];
        __syncthreads();
    }
    if (t == 0) out[0] = sh[0];
}

// ---------------------------------------------------------------------------
extern "C" void kernel_launch(void* const* d_in, const int* in_sizes, int n_in,
                              void* d_out, int out_size, void* d_ws, size_t ws_size,
                              hipStream_t stream) {
    const float* q_single = (const float*)d_in[0];
    const float* d_single = (const float*)d_in[1];
    const float* q_multi  = (const float*)d_in[2];
    const float* d_multi  = (const float*)d_in[3];

    const size_t DN = (size_t)BB * NS * DD;   // 8388608 elems
    const size_t QN = (size_t)BB * NQ * DD;   // 262144 elems

    u16b* dh = (u16b*)d_ws;
    u16b* dl = dh + DN;
    u16b* qh = dl + DN;
    u16b* ql = qh + QN;
    float* scores_mv = (float*)(ql + QN);     // 16B-aligned byte offset
    float* sv_part   = scores_mv + BB * BB;

    convert_kernel<<<4096, 256, 0, stream>>>(q_multi, d_multi, qh, ql, dh, dl);

    dim3 grid(BB, BB);  // x = c, y = b
    pair_kernel<<<grid, 256, 0, stream>>>(qh, ql, dh, dl, scores_mv);

    sv_kernel<<<BB, 256, 0, stream>>>(q_single, d_single, sv_part);

    finalize_kernel<<<1, 64, 0, stream>>>(scores_mv, sv_part, (float*)d_out);
}
